// GroupedExperts_86346022519553
// MI455X (gfx1250) — matrix operand order repaired and checked by reading, not compiled, not run on hardware
//
#include <hip/hip_runtime.h>
#include <hip/hip_bf16.h>
#include <cstdint>

// ---------------------------------------------------------------------------
// Grouped MoE experts (SwiGLU FFN), bf16 WMMA on gfx1250 (MI455X).
//   h   = silu(x @ w1^T) * (x @ w3^T)   per expert   (fused, bf16 -> ws)
//   out = h @ w2^T                       per expert   (f32 -> d_out)
// Pre-convert x/w1/w3/w2 to bf16 in workspace once, then pure bf16 WMMA GEMMs
// with double-buffered LDS fed by GLOBAL_LOAD_ASYNC_TO_LDS_B128 (ASYNCcnt
// pipeline). Fallback instantiations keep inline f32->bf16 conversion if the
// workspace is too small, and a sync copy path if the async builtin is absent.
// ---------------------------------------------------------------------------

#define E_ 8
#define D_ 2048
#define H_ 5632
#define T_ 32768

#define BM  128         // token tile
#define BN  128         // output-column tile
#define BK  64          // K tile (two WMMA K steps)
#define LDK 72          // LDS row stride in bf16 (64 + 8 pad = 144B = 9*16B)
#define NTHREADS 256    // 8 wave32

typedef __bf16 bf16_t;
typedef __bf16 v16bf __attribute__((ext_vector_type(16)));
typedef __bf16 v8bf  __attribute__((ext_vector_type(8)));
typedef float  v8f   __attribute__((ext_vector_type(8)));
typedef int    v4i_vs __attribute__((vector_size(16)));   // builtin's pointee type

#if __has_builtin(__builtin_amdgcn_global_load_async_to_lds_b128)
#define USE_ASYNC_LDS 1
#else
#define USE_ASYNC_LDS 0
#endif

__device__ __forceinline__ unsigned fbits(float x) {
  union { float f; unsigned u; } v; v.f = x; return v.u;
}

// pack two f32 -> packed bf16 pair (low = a, high = b)
__device__ __forceinline__ unsigned packbf(float a, float b) {
#if __has_builtin(__builtin_amdgcn_cvt_pk_bf16_f32)
  auto r = __builtin_amdgcn_cvt_pk_bf16_f32(a, b);
  union { decltype(r) r_; unsigned u; } v; v.r_ = r; return v.u;
#elif __has_builtin(__builtin_amdgcn_perm)
  return __builtin_amdgcn_perm(fbits(b), fbits(a), 0x07060302u);
#else
  return (fbits(a) >> 16) | (fbits(b) & 0xffff0000u);
#endif
}

__device__ __forceinline__ bf16_t to_bf16(float f) {
  unsigned u = fbits(f);
  unsigned r = (u + 0x7FFFu + ((u >> 16) & 1u)) >> 16;  // RNE
  union { unsigned short s; bf16_t b; } v;
  v.s = (unsigned short)r; return v.b;
}

__device__ __forceinline__ v8f zero8() {
  v8f z;
#pragma unroll
  for (int i = 0; i < 8; ++i) z[i] = 0.0f;
  return z;
}

// A fragment (16x32 bf16, MxK): lane L<16 -> row L, K 0..7 & 16..23
//                               lane L>=16 -> row L-16, K 8..15 & 24..31
__device__ __forceinline__ v16bf lds_frag_a(const bf16_t* base, int lane) {
  const int row = lane & 15;
  const int hi  = lane >> 4;
  const bf16_t* r = base + row * LDK;
  union { struct { v8bf lo; v8bf hi; } p; v16bf v; } u;
  u.p.lo = *(const v8bf*)(r + hi * 8);
  u.p.hi = *(const v8bf*)(r + 16 + hi * 8);
  return u.v;
}

// B fragment (32x16 bf16, KxN), LDS row n holds K-contiguous weights:
// lane L<16 -> col L, K 0..7 & 16..23 ; lane L>=16 -> col L-16, K 8..15 & 24..31
__device__ __forceinline__ v16bf lds_frag_b(const bf16_t* base, int lane) {
  const int col = lane & 15;
  const int hi  = lane >> 4;
  const bf16_t* r = base + col * LDK;
  union { struct { v8bf lo; v8bf hi; } p; v16bf v; } u;
  u.p.lo = *(const v8bf*)(r + hi * 8);
  u.p.hi = *(const v8bf*)(r + 16 + hi * 8);
  return u.v;
}

__device__ __forceinline__ v8f wmma_bf16(v16bf a, v16bf b, v8f c) {
  return __builtin_amdgcn_wmma_f32_16x16x32_bf16(
      false, a, false, b, (short)0, c, false, false);
}

__device__ __forceinline__ int expert_of_tile(const int* __restrict__ counts,
                                              int row0) {
  int e = 0, acc = 0;
#pragma unroll
  for (int i = 0; i < E_; ++i) {
    int c = counts[i];
    if (row0 >= acc + c) { acc += c; e = i + 1; }
  }
  return (e < E_) ? e : (E_ - 1);
}

// wait until <= N of this wave's async LDS loads remain outstanding
template <int N>
__device__ __forceinline__ void wait_async() {
#if USE_ASYNC_LDS
#if __has_builtin(__builtin_amdgcn_s_wait_asynccnt)
  __builtin_amdgcn_s_wait_asynccnt((short)N);
#else
  asm volatile("s_wait_asynccnt %0" :: "i"(N) : "memory");
#endif
#endif
}

// ---- tile staging: 128x64 elements into LDS (bf16) ------------------------
// bf16 source: async global->LDS copy (2 x b128 per 16-elem chunk)
__device__ __forceinline__ void issue_chunk(bf16_t* dp, const bf16_t* sp) {
#if USE_ASYNC_LDS
  __builtin_amdgcn_global_load_async_to_lds_b128(
      (v4i_vs*)const_cast<bf16_t*>(sp), (v4i_vs*)dp, 0, 0);
  __builtin_amdgcn_global_load_async_to_lds_b128(
      (v4i_vs*)const_cast<bf16_t*>(sp + 8), (v4i_vs*)(dp + 8), 0, 0);
#else
  *(v8bf*)dp       = *(const v8bf*)sp;
  *(v8bf*)(dp + 8) = *(const v8bf*)(sp + 8);
#endif
}

// f32 source: sync load + pack (fallback path when ws can't hold bf16 copies)
__device__ __forceinline__ void issue_chunk(bf16_t* dp, const float* sp) {
  __builtin_prefetch(sp + BK, 0, 3);
  const float4 f0 = *(const float4*)(sp);
  const float4 f1 = *(const float4*)(sp + 4);
  const float4 f2 = *(const float4*)(sp + 8);
  const float4 f3 = *(const float4*)(sp + 12);
  uint4 q0 = make_uint4(packbf(f0.x, f0.y), packbf(f0.z, f0.w),
                        packbf(f1.x, f1.y), packbf(f1.z, f1.w));
  uint4 q1 = make_uint4(packbf(f2.x, f2.y), packbf(f2.z, f2.w),
                        packbf(f3.x, f3.y), packbf(f3.z, f3.w));
  *(uint4*)dp       = q0;
  *(uint4*)(dp + 8) = q1;
}

template <typename SrcT>
__device__ __forceinline__ void issue_tile128(bf16_t* __restrict__ dst,
                                              const SrcT* __restrict__ src,
                                              size_t ld, int tid) {
#pragma unroll
  for (int i = 0; i < 2; ++i) {
    const int c   = tid + i * NTHREADS;   // 0..511
    const int r   = c >> 2;               // 0..127
    const int col = (c & 3) * 16;         // 0,16,32,48
    issue_chunk(dst + r * LDK + col, src + (size_t)r * ld + col);
  }
}

// async loads in flight per tile per thread (bf16 path: 4 b128; f32 path: 0)
template <typename SrcT>
constexpr int tile_async() { return (sizeof(SrcT) == 2) ? 4 : 0; }

// ---------------------------------------------------------------------------
// Kernel 1 (fused gate+up): h = silu(x@w1^T) * (x@w3^T) -> bf16 hbuf
// grid = (H/BN, T/BM); waves: 2 over M (64 rows), 4 over N (32 cols)
// Double-buffered LDS, async-prefetch pipeline.
// ---------------------------------------------------------------------------
template <typename TA, typename TB>
__global__ __launch_bounds__(NTHREADS)
void moe_gateup_kernel(const TA* __restrict__ xA,
                       const TB* __restrict__ w1,
                       const TB* __restrict__ w3,
                       const int* __restrict__ counts,
                       bf16_t* __restrict__ hbuf) {
  __shared__ bf16_t sA [2][BM * LDK];
  __shared__ bf16_t sB1[2][BN * LDK];
  __shared__ bf16_t sB3[2][BN * LDK];

  const int row0 = blockIdx.y * BM;
  const int col0 = blockIdx.x * BN;
  const int e = expert_of_tile(counts, row0);
  const TA* __restrict__ xR  = xA + (size_t)row0 * D_;
  const TB* __restrict__ w1e = w1 + (size_t)e * H_ * D_ + (size_t)col0 * D_;
  const TB* __restrict__ w3e = w3 + (size_t)e * H_ * D_ + (size_t)col0 * D_;

  const int tid  = threadIdx.x;
  const int lane = tid & 31;
  const int wave = tid >> 5;
  const int wm   = wave & 1;   // 2 waves * 64 rows
  const int wn   = wave >> 1;  // 4 waves * 32 cols

  constexpr int PT = tile_async<TA>() + 2 * tile_async<TB>();

  v8f accg[4][2], accu[4][2];
#pragma unroll
  for (int i = 0; i < 4; ++i)
#pragma unroll
    for (int j = 0; j < 2; ++j) { accg[i][j] = zero8(); accu[i][j] = zero8(); }

  auto issue = [&](int k0, int b) {
    issue_tile128(sA [b], xR  + k0, (size_t)D_, tid);
    issue_tile128(sB1[b], w1e + k0, (size_t)D_, tid);
    issue_tile128(sB3[b], w3e + k0, (size_t)D_, tid);
  };

  issue(0, 0);
  int buf = 0;
  for (int k0 = 0; k0 < D_; k0 += BK) {
    if (k0 + BK < D_) { issue(k0 + BK, buf ^ 1); wait_async<PT>(); }
    else              { wait_async<0>(); }
    __syncthreads();

#pragma unroll
    for (int ks = 0; ks < 2; ++ks) {
      const bf16_t* aB  = sA [buf] + (wm * 64) * LDK + ks * 32;
      const bf16_t* b1B = sB1[buf] + (wn * 32) * LDK + ks * 32;
      const bf16_t* b3B = sB3[buf] + (wn * 32) * LDK + ks * 32;
      v16bf a[4], b1[2], b3[2];
#pragma unroll
      for (int mi = 0; mi < 4; ++mi) a[mi] = lds_frag_a(aB + mi * 16 * LDK, lane);
#pragma unroll
      for (int nj = 0; nj < 2; ++nj) {
        b1[nj] = lds_frag_b(b1B + nj * 16 * LDK, lane);
        b3[nj] = lds_frag_b(b3B + nj * 16 * LDK, lane);
      }
#pragma unroll
      for (int mi = 0; mi < 4; ++mi)
#pragma unroll
        for (int nj = 0; nj < 2; ++nj) {
          accg[mi][nj] = wmma_bf16(a[mi], b1[nj], accg[mi][nj]);
          accu[mi][nj] = wmma_bf16(a[mi], b3[nj], accu[mi][nj]);
        }
    }
    __syncthreads();
    buf ^= 1;
  }

  const int mo = (lane >> 4) * 8;
  const int n  = lane & 15;
#pragma unroll
  for (int mi = 0; mi < 4; ++mi)
#pragma unroll
    for (int nj = 0; nj < 2; ++nj)
#pragma unroll
      for (int r = 0; r < 8; ++r) {
        const float g = accg[mi][nj][r];
        const float u = accu[mi][nj][r];
        const float h = (g / (1.0f + __expf(-g))) * u;
        const int mm = row0 + wm * 64 + mi * 16 + mo + r;
        const int nn = col0 + wn * 32 + nj * 16 + n;
        hbuf[(size_t)mm * H_ + nn] = to_bf16(h);
      }
}

// ---------------------------------------------------------------------------
// Kernel 2 (down): out = h @ w2^T -> f32
// grid = (D/BN, T/BM)
// ---------------------------------------------------------------------------
template <typename TB>
__global__ __launch_bounds__(NTHREADS)
void moe_down_kernel(const bf16_t* __restrict__ hbuf,
                     const TB* __restrict__ w2,
                     const int* __restrict__ counts,
                     float* __restrict__ out) {
  __shared__ bf16_t sA[2][BM * LDK];
  __shared__ bf16_t sB[2][BN * LDK];

  const int row0 = blockIdx.y * BM;
  const int col0 = blockIdx.x * BN;
  const int e = expert_of_tile(counts, row0);
  const bf16_t* __restrict__ hR  = hbuf + (size_t)row0 * H_;
  const TB* __restrict__     w2e = w2 + (size_t)e * D_ * H_ + (size_t)col0 * H_;

  const int tid  = threadIdx.x;
  const int lane = tid & 31;
  const int wave = tid >> 5;
  const int wm   = wave & 1;
  const int wn   = wave >> 1;

  constexpr int PT = tile_async<bf16_t>() + tile_async<TB>();

  v8f acc[4][2];
#pragma unroll
  for (int i = 0; i < 4; ++i)
#pragma unroll
    for (int j = 0; j < 2; ++j) acc[i][j] = zero8();

  auto issue = [&](int k0, int b) {
    issue_tile128(sA[b], hR  + k0, (size_t)H_, tid);
    issue_tile128(sB[b], w2e + k0, (size_t)H_, tid);
  };

  issue(0, 0);
  int buf = 0;
  for (int k0 = 0; k0 < H_; k0 += BK) {
    if (k0 + BK < H_) { issue(k0 + BK, buf ^ 1); wait_async<PT>(); }
    else              { wait_async<0>(); }
    __syncthreads();

#pragma unroll
    for (int ks = 0; ks < 2; ++ks) {
      const bf16_t* aB = sA[buf] + (wm * 64) * LDK + ks * 32;
      const bf16_t* bB = sB[buf] + (wn * 32) * LDK + ks * 32;
      v16bf a[4], b[2];
#pragma unroll
      for (int mi = 0; mi < 4; ++mi) a[mi] = lds_frag_a(aB + mi * 16 * LDK, lane);
#pragma unroll
      for (int nj = 0; nj < 2; ++nj) b[nj] = lds_frag_b(bB + nj * 16 * LDK, lane);
#pragma unroll
      for (int mi = 0; mi < 4; ++mi)
#pragma unroll
        for (int nj = 0; nj < 2; ++nj)
          acc[mi][nj] = wmma_bf16(a[mi], b[nj], acc[mi][nj]);
    }
    __syncthreads();
    buf ^= 1;
  }

  const int mo = (lane >> 4) * 8;
  const int n  = lane & 15;
#pragma unroll
  for (int mi = 0; mi < 4; ++mi)
#pragma unroll
    for (int nj = 0; nj < 2; ++nj)
#pragma unroll
      for (int r = 0; r < 8; ++r) {
        const int mm = row0 + wm * 64 + mi * 16 + mo + r;
        const int nn = col0 + wn * 32 + nj * 16 + n;
        out[(size_t)mm * D_ + nn] = acc[mi][nj][r];
      }
}

// ---------------------------------------------------------------------------
// f32 -> bf16 bulk conversion (grid-stride, float4 granularity)
// ---------------------------------------------------------------------------
__global__ __launch_bounds__(NTHREADS)
void cvt_f32_to_bf16_kernel(const float* __restrict__ src,
                            bf16_t* __restrict__ dst, long n4) {
  long i = (long)blockIdx.x * blockDim.x + threadIdx.x;
  const long stride = (long)gridDim.x * blockDim.x;
  for (; i < n4; i += stride) {
    const float4 f = ((const float4*)src)[i];
    ((uint2*)dst)[i] = make_uint2(packbf(f.x, f.y), packbf(f.z, f.w));
  }
}

// ---------------------------------------------------------------------------
extern "C" void kernel_launch(void* const* d_in, const int* in_sizes, int n_in,
                              void* d_out, int out_size, void* d_ws, size_t ws_size,
                              hipStream_t stream) {
  (void)in_sizes; (void)n_in; (void)out_size;
  const float* x      = (const float*)d_in[0];
  const float* w1     = (const float*)d_in[1];
  const float* w2     = (const float*)d_in[2];
  const float* w3     = (const float*)d_in[3];
  const int*   counts = (const int*)d_in[4];
  float* out = (float*)d_out;

  const size_t nH   = (size_t)T_ * H_;          // h elements
  const size_t nX   = (size_t)T_ * D_;          // x elements
  const size_t nW   = (size_t)E_ * H_ * D_;     // per-weight elements
  const size_t need = (nH + nX + 3 * nW) * sizeof(bf16_t);

  bf16_t* hbuf = (bf16_t*)d_ws;
  dim3 block(NTHREADS);
  dim3 g1(H_ / BN, T_ / BM);   // (44, 256)
  dim3 g2(D_ / BN, T_ / BM);   // (16, 256)

  if (ws_size >= need) {
    bf16_t* xb  = hbuf + nH;
    bf16_t* w1b = xb + nX;
    bf16_t* w3b = w1b + nW;
    bf16_t* w2b = w3b + nW;
    dim3 cg(8192);
    cvt_f32_to_bf16_kernel<<<cg, block, 0, stream>>>(x,  xb,  (long)(nX / 4));
    cvt_f32_to_bf16_kernel<<<cg, block, 0, stream>>>(w1, w1b, (long)(nW / 4));
    cvt_f32_to_bf16_kernel<<<cg, block, 0, stream>>>(w3, w3b, (long)(nW / 4));
    cvt_f32_to_bf16_kernel<<<cg, block, 0, stream>>>(w2, w2b, (long)(nW / 4));
    moe_gateup_kernel<bf16_t, bf16_t><<<g1, block, 0, stream>>>(
        xb, w1b, w3b, counts, hbuf);
    moe_down_kernel<bf16_t><<<g2, block, 0, stream>>>(hbuf, w2b, counts, out);
  } else {
    // workspace only fits h: convert on the fly inside the GEMMs
    moe_gateup_kernel<float, float><<<g1, block, 0, stream>>>(
        x, w1, w3, counts, hbuf);
    moe_down_kernel<float><<<g2, block, 0, stream>>>(hbuf, w2, counts, out);
  }
}
